// Linear_80874234183916
// MI455X (gfx1250) — matrix operand order repaired and checked by reading, not compiled, not run on hardware
//
#include <hip/hip_runtime.h>
#include <hip/hip_bf16.h>

// ---------------------------------------------------------------------------
// Approximate Linear: y = x @ mask(w).T + bias
//   mask keeps |w| >= thr where thr = ascending-rank-r element of |w|,
//   r = n - max_iter = 8388608 (n = 4096*4096, max_iter = n/2).
// Pipeline:
//   1) zero histogram workspace
//   2) radix histogram (high 16 bits of |w| bit pattern)
//   3) select high bucket + residual rank
//   4) radix histogram (low 16 bits, restricted to selected bucket)
//   5) select exact threshold
//   6) tiled WMMA GEMM (bf16x3 split for ~fp32 accuracy) with mask fused
//      into the weight -> LDS staging, + bias epilogue
// ---------------------------------------------------------------------------

typedef __attribute__((ext_vector_type(16))) __bf16 v16bf;
typedef __attribute__((ext_vector_type(8)))  __bf16 v8bf;
typedef __attribute__((ext_vector_type(8)))  float  v8f;

#define N_TOK 8192
#define OUT_F 4096
#define IN_F  4096
#define W_N   (4096LL * 4096LL)
#define RANK_R 8388608ULL   // n - max_iter

#define BM 128
#define BN 128
#define BK 32
#define LDA 40              // padded LDS row stride (bf16 elems): conflict-free

union FragBF { v16bf v; v8bf h[2]; };

// ------------------------------ selection ----------------------------------

__global__ void zero_ws_kernel(unsigned int* ws, int n) {
    int i = blockIdx.x * blockDim.x + threadIdx.x;
    if (i < n) ws[i] = 0u;
}

__global__ void hist_hi_kernel(const float* __restrict__ w,
                               unsigned int* __restrict__ hist) {
    long long i = (long long)blockIdx.x * blockDim.x + threadIdx.x;
    long long stride = (long long)gridDim.x * blockDim.x;
    for (; i < W_N; i += stride) {
        unsigned int bits = __float_as_uint(w[i]) & 0x7FFFFFFFu;
        atomicAdd(&hist[bits >> 16], 1u);
    }
}

__global__ void select_hi_kernel(const unsigned int* __restrict__ hist,
                                 unsigned int* __restrict__ outB,
                                 unsigned int* __restrict__ outR2) {
    __shared__ unsigned int partial[256];
    int t = threadIdx.x;
    unsigned int s = 0;
    for (int b = 0; b < 256; ++b) s += hist[t * 256 + b];
    partial[t] = s;
    __syncthreads();
    if (t == 0) {
        unsigned long long cum = 0;
        for (int seg = 0; seg < 256; ++seg) {
            if (cum + partial[seg] > RANK_R) {
                unsigned long long below = cum;
                for (int b = seg * 256;; ++b) {
                    unsigned int c = hist[b];
                    if (below + c > RANK_R) {
                        *outB  = (unsigned int)b;
                        *outR2 = (unsigned int)(RANK_R - below);
                        break;
                    }
                    below += c;
                }
                break;
            }
            cum += partial[seg];
        }
    }
}

__global__ void hist_lo_kernel(const float* __restrict__ w,
                               unsigned int* __restrict__ hist,
                               const unsigned int* __restrict__ selB) {
    const unsigned int B = *selB;
    long long i = (long long)blockIdx.x * blockDim.x + threadIdx.x;
    long long stride = (long long)gridDim.x * blockDim.x;
    for (; i < W_N; i += stride) {
        unsigned int bits = __float_as_uint(w[i]) & 0x7FFFFFFFu;
        if ((bits >> 16) == B) atomicAdd(&hist[bits & 0xFFFFu], 1u);
    }
}

__global__ void select_lo_kernel(const unsigned int* __restrict__ hist,
                                 const unsigned int* __restrict__ selR2,
                                 const unsigned int* __restrict__ selB,
                                 float* __restrict__ thrOut) {
    __shared__ unsigned int partial[256];
    int t = threadIdx.x;
    unsigned int s = 0;
    for (int b = 0; b < 256; ++b) s += hist[t * 256 + b];
    partial[t] = s;
    __syncthreads();
    if (t == 0) {
        const unsigned long long r2 = *selR2;
        const unsigned int B = *selB;
        unsigned long long cum = 0;
        for (int seg = 0; seg < 256; ++seg) {
            if (cum + partial[seg] > r2) {
                unsigned long long below = cum;
                for (int b = seg * 256;; ++b) {
                    unsigned int c = hist[b];
                    if (below + c > r2) {
                        unsigned int key = (B << 16) | (unsigned int)b;
                        *thrOut = __uint_as_float(key);
                        break;
                    }
                    below += c;
                }
                break;
            }
            cum += partial[seg];
        }
    }
}

// ------------------------------ WMMA GEMM ----------------------------------

__global__ __launch_bounds__(256)
void approx_linear_gemm(const float* __restrict__ x,
                        const float* __restrict__ w,
                        const float* __restrict__ bias,
                        const float* __restrict__ thr_p,
                        float* __restrict__ out) {
    __shared__ __bf16 sAhi[BM * LDA];
    __shared__ __bf16 sAlo[BM * LDA];
    __shared__ __bf16 sBhi[BN * LDA];
    __shared__ __bf16 sBlo[BN * LDA];

    const float thr = *thr_p;
    const int tid   = threadIdx.x;
    const int lane  = tid & 31;
    const int wid   = tid >> 5;
    const int waveM = wid >> 2;        // 0..1  -> 64-row patch
    const int waveN = wid & 3;         // 0..3  -> 32-col patch
    const int bmBase = blockIdx.y * BM;
    const int bnBase = blockIdx.x * BN;

    const float4* __restrict__ xg = reinterpret_cast<const float4*>(x);
    const float4* __restrict__ wg = reinterpret_cast<const float4*>(w);

    float4 regA[4], regW[4];

    auto loadTiles = [&](int k0) {
#pragma unroll
        for (int i = 0; i < 4; ++i) {
            int q   = i * 256 + tid;     // 0..1023 : 128 rows x 8 float4
            int row = q >> 3;
            int c4  = q & 7;
            size_t koff = (size_t)(k0 >> 2) + c4;
            regA[i] = xg[(size_t)(bmBase + row) * (IN_F / 4) + koff];
            regW[i] = wg[(size_t)(bnBase + row) * (IN_F / 4) + koff];
        }
    };

    auto stageTiles = [&]() {
#pragma unroll
        for (int i = 0; i < 4; ++i) {
            int q   = i * 256 + tid;
            int row = q >> 3;
            int kc  = (q & 7) * 4;
            const float* ax = reinterpret_cast<const float*>(&regA[i]);
            const float* aw = reinterpret_cast<const float*>(&regW[i]);
#pragma unroll
            for (int j = 0; j < 4; ++j) {
                float xv = ax[j];
                __bf16 xh = (__bf16)xv;
                sAhi[row * LDA + kc + j] = xh;
                sAlo[row * LDA + kc + j] = (__bf16)(xv - (float)xh);
                float wv = aw[j];
                wv = (__builtin_fabsf(wv) >= thr) ? wv : 0.0f;  // fused mask
                __bf16 wh = (__bf16)wv;
                sBhi[row * LDA + kc + j] = wh;
                sBlo[row * LDA + kc + j] = (__bf16)(wv - (float)wh);
            }
        }
    };

    v8f acc[4][2];
#pragma unroll
    for (int ms = 0; ms < 4; ++ms)
#pragma unroll
        for (int ns = 0; ns < 2; ++ns)
            acc[ms][ns] = (v8f){0.f, 0.f, 0.f, 0.f, 0.f, 0.f, 0.f, 0.f};

    const int lh = lane >> 4;   // lane half
    const int ll = lane & 15;

    auto computeStep = [&]() {
        // B fragments: lane holds column N = lane%16, K = (lane/16)*16 + j
        FragBF bh[2], bl[2];
#pragma unroll
        for (int ns = 0; ns < 2; ++ns) {
            int cb = (waveN * 32 + ns * 16 + ll) * LDA + lh * 16;
            bh[ns].h[0] = *reinterpret_cast<const v8bf*>(&sBhi[cb]);
            bh[ns].h[1] = *reinterpret_cast<const v8bf*>(&sBhi[cb + 8]);
            bl[ns].h[0] = *reinterpret_cast<const v8bf*>(&sBlo[cb]);
            bl[ns].h[1] = *reinterpret_cast<const v8bf*>(&sBlo[cb + 8]);
        }
#pragma unroll
        for (int ms = 0; ms < 4; ++ms) {
            // A fragments: row M = lane%16; K = (lane/16)*16 + j
            int rb = (waveM * 64 + ms * 16 + ll) * LDA + lh * 16;
            FragBF ah, al;
            ah.h[0] = *reinterpret_cast<const v8bf*>(&sAhi[rb]);
            ah.h[1] = *reinterpret_cast<const v8bf*>(&sAhi[rb + 8]);
            al.h[0] = *reinterpret_cast<const v8bf*>(&sAlo[rb]);
            al.h[1] = *reinterpret_cast<const v8bf*>(&sAlo[rb + 8]);
#pragma unroll
            for (int ns = 0; ns < 2; ++ns) {
                acc[ms][ns] = __builtin_amdgcn_wmma_f32_16x16x32_bf16(
                    false, ah.v, false, bh[ns].v, (short)0, acc[ms][ns], false, false);
                acc[ms][ns] = __builtin_amdgcn_wmma_f32_16x16x32_bf16(
                    false, ah.v, false, bl[ns].v, (short)0, acc[ms][ns], false, false);
                acc[ms][ns] = __builtin_amdgcn_wmma_f32_16x16x32_bf16(
                    false, al.v, false, bh[ns].v, (short)0, acc[ms][ns], false, false);
            }
        }
    };

    loadTiles(0);
    constexpr int NK = IN_F / BK;   // 128
    for (int kt = 0; kt < NK; ++kt) {
        __syncthreads();
        stageTiles();
        __syncthreads();
        if (kt + 1 < NK) loadTiles((kt + 1) * BK);   // overlap VMEM with WMMA
        computeStep();
    }

    // Epilogue: C/D layout -> lane L: N = L%16, VGPR r: M = r + 8*(L/16)
#pragma unroll
    for (int ms = 0; ms < 4; ++ms) {
        int gRow = bmBase + waveM * 64 + ms * 16 + lh * 8;
#pragma unroll
        for (int ns = 0; ns < 2; ++ns) {
            int gCol = bnBase + waveN * 32 + ns * 16 + ll;
            float bv = bias[gCol];
#pragma unroll
            for (int r = 0; r < 8; ++r) {
                out[(size_t)(gRow + r) * OUT_F + gCol] = acc[ms][ns][r] + bv;
            }
        }
    }
}

// ------------------------------ launcher -----------------------------------

extern "C" void kernel_launch(void* const* d_in, const int* in_sizes, int n_in,
                              void* d_out, int out_size, void* d_ws, size_t ws_size,
                              hipStream_t stream) {
    const float* x    = (const float*)d_in[0];
    const float* w    = (const float*)d_in[1];
    const float* bias = (const float*)d_in[2];
    float* out        = (float*)d_out;

    unsigned int* ws     = (unsigned int*)d_ws;
    unsigned int* histHi = ws;             // 65536 uints
    unsigned int* histLo = ws + 65536;     // 65536 uints
    unsigned int* selB   = ws + 131072;
    unsigned int* selR2  = ws + 131073;
    float*        thr    = (float*)(ws + 131074);
    const int wsWords = 131075;

    zero_ws_kernel<<<(wsWords + 255) / 256, 256, 0, stream>>>(ws, wsWords);
    hist_hi_kernel<<<2048, 256, 0, stream>>>(w, histHi);
    select_hi_kernel<<<1, 256, 0, stream>>>(histHi, selB, selR2);
    hist_lo_kernel<<<2048, 256, 0, stream>>>(w, histLo, selB);
    select_lo_kernel<<<1, 256, 0, stream>>>(histLo, selR2, selB, thr);

    dim3 grid(OUT_F / BN, N_TOK / BM);   // (32, 64)
    approx_linear_gemm<<<grid, 256, 0, stream>>>(x, w, bias, thr, out);
}
